// ChebLocalModel_30923764531696
// MI455X (gfx1250) — matrix operand order repaired and checked by reading, not compiled, not run on hardware
//
#include <hip/hip_runtime.h>
#include <hip/hip_bf16.h>

#define NN 20000
#define EE 320000

typedef __attribute__((ext_vector_type(16))) __bf16 v16bf;
typedef __attribute__((ext_vector_type(8)))  float  v8f;

struct alignas(16) US8 { unsigned short v[8]; };
struct alignas(8)  BF4 { __bf16 v[4]; };
struct alignas(4)  BF2 { __bf16 v[2]; };
union FragBF { v16bf v; unsigned short u[16]; };

// ---------------------------------------------------------------------------
// Utility elementwise kernels
// ---------------------------------------------------------------------------
__global__ void zero_f32(float* __restrict__ p, long long n) {
  long long i = (long long)blockIdx.x * blockDim.x + threadIdx.x;
  if (i < n) p[i] = 0.0f;
}

__global__ void scale_copy(float* __restrict__ dst, const float* __restrict__ src,
                           float alpha, long long n) {
  long long i = (long long)blockIdx.x * blockDim.x + threadIdx.x;
  if (i < n) dst[i] = alpha * src[i];
}

// ---------------------------------------------------------------------------
// Graph preprocessing: degree, d^{-1/2}, per-edge Laplacian weight
// ---------------------------------------------------------------------------
__global__ void deg_kernel(const int* __restrict__ row, float* __restrict__ deg, int e) {
  int i = blockIdx.x * blockDim.x + threadIdx.x;
  if (i < e) atomicAdd(&deg[row[i]], 1.0f);
}

__global__ void dinv_kernel(const float* __restrict__ deg, float* __restrict__ dinv, int n) {
  int i = blockIdx.x * blockDim.x + threadIdx.x;
  if (i < n) {
    float d = deg[i];
    dinv[i] = (d > 0.0f) ? rsqrtf(fmaxf(d, 1.0f)) : 0.0f;
  }
}

__global__ void norm_kernel(const int* __restrict__ row, const int* __restrict__ col,
                            const float* __restrict__ dinv, float* __restrict__ nrm, int e) {
  int i = blockIdx.x * blockDim.x + threadIdx.x;
  if (i < e) nrm[i] = -dinv[row[i]] * dinv[col[i]];
}

// out[col] += scale * norm[e] * h[row]  (4 features / thread, vectorized load)
__global__ void prop_scatter(const float* __restrict__ h, const int* __restrict__ row,
                             const int* __restrict__ col, const float* __restrict__ nrm,
                             float scale, float* __restrict__ out, int e, int F) {
  int F4 = F >> 2;
  long long tid = (long long)blockIdx.x * blockDim.x + threadIdx.x;
  if (tid >= (long long)e * F4) return;
  int ed = (int)(tid / F4);
  int f  = ((int)(tid % F4)) << 2;
  int r = row[ed], c = col[ed];
  float w = scale * nrm[ed];
  const float4 hv = *(const float4*)(h + (size_t)r * F + f);
  float* o = out + (size_t)c * F + f;
  atomicAdd(o + 0, w * hv.x);
  atomicAdd(o + 1, w * hv.y);
  atomicAdd(o + 2, w * hv.z);
  atomicAdd(o + 3, w * hv.w);
}

// ---------------------------------------------------------------------------
// WMMA GEMM: C[M,Nc] (+)= A[M,K] @ B[K,Nc], fp32 I/O, bf16 matrix cores.
// 256 threads = 8 waves; block tile 128x64, BK=32; wave tile 32x32
// (4 wmma / K-step). Register-pipelined: tile k+1 global loads are issued
// before the WMMA work of tile k so HBM latency overlaps matrix math.
// ---------------------------------------------------------------------------
#define BM 128
#define BN 64
#define BK 32
#define LDT 40   // padded LDS row stride (bf16 elems): 80B rows, 16B-aligned

__global__ __launch_bounds__(256, 2)
void gemm_wmma_bf16(const float* __restrict__ A, const float* __restrict__ B,
                    float* __restrict__ C, int M, int K, int Nc, int accumulate) {
  __shared__ __bf16 As[BM * LDT];   // As[m][k]
  __shared__ __bf16 Bs[BN * LDT];   // Bs[n][k] (transposed on store)

  const int t    = threadIdx.x;
  const int bm   = blockIdx.x * BM;
  const int bn   = blockIdx.y * BN;
  const int wave = t >> 5;
  const int lane = t & 31;
  const int wm   = (wave & 3) << 5;   // 0,32,64,96  (32 rows per wave)
  const int wn   = (wave >> 2) << 5;  // 0,32        (32 cols per wave)
  const int hi   = lane >> 4;         // lane group
  const int mr   = lane & 15;

  // B staging coords: each thread owns k-rows {kk2, kk2+1} x 4 n-columns
  const int kk2 = (t >> 4) << 1;      // 0,2,...,30
  const int n4  = t & 15;             // float4 column -> n = n4*4..+3

  v8f acc00 = {}, acc01 = {}, acc10 = {}, acc11 = {};

  float4 ra[4], rb0, rb1;

  // register-resident tile load (no LDS deps -> loads batch, waits coalesce)
  auto load_tiles = [&](int k0) {
#pragma unroll
    for (int i = 0; i < 4; ++i) {
      int l  = i * 256 + t;          // float4-granule index, 1024 total
      int r  = l >> 3;               // row 0..127
      int c4 = l & 7;                // float4 column 0..7
      int gr = bm + r;
      gr = gr < M ? gr : (M - 1);    // clamp -> unconditional load; OOB rows
                                     // only feed OOB outputs (never stored)
      ra[i] = *(const float4*)(A + (size_t)gr * K + k0 + (c4 << 2));
    }
    rb0 = *(const float4*)(B + (size_t)(k0 + kk2)     * Nc + bn + (n4 << 2));
    rb1 = *(const float4*)(B + (size_t)(k0 + kk2 + 1) * Nc + bn + (n4 << 2));
  };

  auto store_tiles = [&]() {
#pragma unroll
    for (int i = 0; i < 4; ++i) {
      int l  = i * 256 + t;
      int r  = l >> 3;
      int c4 = l & 7;
      BF4 pk;
      pk.v[0] = (__bf16)ra[i].x;
      pk.v[1] = (__bf16)ra[i].y;
      pk.v[2] = (__bf16)ra[i].z;
      pk.v[3] = (__bf16)ra[i].w;
      *(BF4*)(As + r * LDT + (c4 << 2)) = pk;          // ds_store_b64
    }
    const float b0[4] = {rb0.x, rb0.y, rb0.z, rb0.w};
    const float b1[4] = {rb1.x, rb1.y, rb1.z, rb1.w};
#pragma unroll
    for (int j = 0; j < 4; ++j) {
      BF2 p;                                           // (k=kk2, k=kk2+1)
      p.v[0] = (__bf16)b0[j];
      p.v[1] = (__bf16)b1[j];
      *(BF2*)(Bs + ((n4 << 2) + j) * LDT + kk2) = p;   // ds_store_b32
    }
  };

  load_tiles(0);

  for (int k0 = 0; k0 < K; k0 += BK) {
    __syncthreads();        // previous frag reads done (no-op first iter)
    store_tiles();
    __syncthreads();        // tiles visible
    if (k0 + BK < K) load_tiles(k0 + BK);   // issue next loads before math

    // A fragments: lane holds M=lane%16; e<8 -> k=e+8*hi, e>=8 -> k=16+(e-8)+8*hi
    const US8* Ar0 = (const US8*)(As + (wm + mr) * LDT);
    const US8* Ar1 = (const US8*)(As + (wm + 16 + mr) * LDT);
    US8 a0l = Ar0[hi], a0h = Ar0[2 + hi];
    US8 a1l = Ar1[hi], a1h = Ar1[2 + hi];
    FragBF fa0, fa1;
#pragma unroll
    for (int e2 = 0; e2 < 8; ++e2) {
      fa0.u[e2] = a0l.v[e2]; fa0.u[8 + e2] = a0h.v[e2];
      fa1.u[e2] = a1l.v[e2]; fa1.u[8 + e2] = a1h.v[e2];
    }

    // B fragments: lane holds N=lane%16; e<8 -> k=e+8*hi, e>=8 -> k=16+(e-8)+8*hi
    const US8* Br0 = (const US8*)(Bs + (wn + mr) * LDT);
    const US8* Br1 = (const US8*)(Bs + (wn + 16 + mr) * LDT);
    US8 b0l = Br0[hi], b0h = Br0[2 + hi];
    US8 b1l = Br1[hi], b1h = Br1[2 + hi];
    FragBF fb0, fb1;
#pragma unroll
    for (int e2 = 0; e2 < 8; ++e2) {
      fb0.u[e2] = b0l.v[e2]; fb0.u[8 + e2] = b0h.v[e2];
      fb1.u[e2] = b1l.v[e2]; fb1.u[8 + e2] = b1h.v[e2];
    }

    acc00 = __builtin_amdgcn_wmma_f32_16x16x32_bf16(false, fa0.v, false, fb0.v,
                                                    (short)0, acc00, false, false);
    acc01 = __builtin_amdgcn_wmma_f32_16x16x32_bf16(false, fa0.v, false, fb1.v,
                                                    (short)0, acc01, false, false);
    acc10 = __builtin_amdgcn_wmma_f32_16x16x32_bf16(false, fa1.v, false, fb0.v,
                                                    (short)0, acc10, false, false);
    acc11 = __builtin_amdgcn_wmma_f32_16x16x32_bf16(false, fa1.v, false, fb1.v,
                                                    (short)0, acc11, false, false);
  }

  // C layout: VGPR r -> M = r + 8*hi, N = lane%16
#pragma unroll
  for (int r = 0; r < 8; ++r) {
    int gm0 = bm + wm + 8 * hi + r;
    int gm1 = gm0 + 16;
    if (gm0 < M) {
      size_t i0 = (size_t)gm0 * Nc + bn + wn + mr;
      float v0 = acc00[r], v1 = acc01[r];
      if (accumulate) { v0 += C[i0]; v1 += C[i0 + 16]; }
      C[i0] = v0; C[i0 + 16] = v1;
    }
    if (gm1 < M) {
      size_t i1 = (size_t)gm1 * Nc + bn + wn + mr;
      float v2 = acc10[r], v3 = acc11[r];
      if (accumulate) { v2 += C[i1]; v3 += C[i1 + 16]; }
      C[i1] = v2; C[i1 + 16] = v3;
    }
  }
}

// ---------------------------------------------------------------------------
// Fused epilogue: v = relu(acc + res + cheb_b + res_b); out = LN(v)*g + b
// one 256-thread block per node row; wave32 shuffle reduction
// ---------------------------------------------------------------------------
__global__ __launch_bounds__(256)
void epilogue_ln(const float* __restrict__ acc, const float* __restrict__ res,
                 const float* __restrict__ cb, const float* __restrict__ rb,
                 const float* __restrict__ g, const float* __restrict__ be,
                 float* __restrict__ out, int F) {
  int row = blockIdx.x;
  int t = threadIdx.x;
  int per = F >> 8;         // F/256: 1, 2 or 4
  float vals[4];
  float s = 0.0f, s2 = 0.0f;
#pragma unroll 4
  for (int i = 0; i < 4; ++i) {
    if (i < per) {
      int f = t + (i << 8);
      float v = acc[(size_t)row * F + f] + res[(size_t)row * F + f] + cb[f] + rb[f];
      v = fmaxf(v, 0.0f);
      vals[i] = v; s += v; s2 += v * v;
    }
  }
#pragma unroll
  for (int o = 16; o > 0; o >>= 1) {
    s  += __shfl_xor(s,  o, 32);
    s2 += __shfl_xor(s2, o, 32);
  }
  __shared__ float sh[16];
  int wave = t >> 5, lane = t & 31;
  if (lane == 0) { sh[wave] = s; sh[8 + wave] = s2; }
  __syncthreads();
  s = 0.0f; s2 = 0.0f;
#pragma unroll
  for (int w = 0; w < 8; ++w) { s += sh[w]; s2 += sh[8 + w]; }
  float mu   = s / (float)F;
  float var  = s2 / (float)F - mu * mu;
  float rinv = rsqrtf(var + 1e-5f);
#pragma unroll 4
  for (int i = 0; i < 4; ++i) {
    if (i < per) {
      int f = t + (i << 8);
      out[(size_t)row * F + f] = (vals[i] - mu) * rinv * g[f] + be[f];
    }
  }
}

// ---------------------------------------------------------------------------
// Host orchestration
// ---------------------------------------------------------------------------
static inline void launch_zero(float* p, long long n, hipStream_t s) {
  zero_f32<<<(unsigned)((n + 255) / 256), 256, 0, s>>>(p, n);
}
static inline void launch_gemm(const float* A, const float* B, float* C,
                               int M, int K, int Nc, int acc, hipStream_t s) {
  dim3 grid((M + BM - 1) / BM, Nc / BN);
  gemm_wmma_bf16<<<grid, 256, 0, s>>>(A, B, C, M, K, Nc, acc);
}
static inline void launch_scatter(const float* h, const int* row, const int* col,
                                  const float* nrm, float scale, float* out,
                                  int F, hipStream_t s) {
  long long total = (long long)EE * (F >> 2);
  prop_scatter<<<(unsigned)((total + 255) / 256), 256, 0, s>>>(h, row, col, nrm,
                                                              scale, out, EE, F);
}

static void run_block(const float* hin, int Fin, int Fout,
                      const float* chebW, const float* chebB,
                      const float* resW, const float* resB,
                      const float* g, const float* b, float* hout,
                      const int* row, const int* col, const float* nrm,
                      float* res, float* accb, float* tA, float* tB,
                      hipStream_t s) {
  // residual: res = hin @ resW
  launch_gemm(hin, resW, res, NN, Fin, Fout, 0, s);
  // T0 term: acc = hin @ W0
  launch_gemm(hin, chebW, accb, NN, Fin, Fout, 0, s);
  // T1 = Lhat @ hin
  launch_zero(tA, (long long)NN * Fin, s);
  launch_scatter(hin, row, col, nrm, 1.0f, tA, Fin, s);
  launch_gemm(tA, chebW + (size_t)Fin * Fout, accb, NN, Fin, Fout, 1, s);
  // T2 = 2*Lhat@T1 - T0  (init tB = -hin, scatter with weight 2)
  scale_copy<<<(unsigned)(((long long)NN * Fin + 255) / 256), 256, 0, s>>>(
      tB, hin, -1.0f, (long long)NN * Fin);
  launch_scatter(tA, row, col, nrm, 2.0f, tB, Fin, s);
  launch_gemm(tB, chebW + 2 * (size_t)Fin * Fout, accb, NN, Fin, Fout, 1, s);
  // epilogue: LN(relu(acc + cheb_b + res + res_b))
  epilogue_ln<<<NN, 256, 0, s>>>(accb, res, chebB, resB, g, b, hout, Fout);
}

extern "C" void kernel_launch(void* const* d_in, const int* in_sizes, int n_in,
                              void* d_out, int out_size, void* d_ws, size_t ws_size,
                              hipStream_t stream) {
  const float* x       = (const float*)d_in[0];
  const int*   edge    = (const int*)d_in[1];     // [2, E]
  const float* cheb1_w = (const float*)d_in[2];
  const float* cheb1_b = (const float*)d_in[3];
  const float* cheb2_w = (const float*)d_in[4];
  const float* cheb2_b = (const float*)d_in[5];
  const float* cheb3_w = (const float*)d_in[6];
  const float* cheb3_b = (const float*)d_in[7];
  const float* res1_w  = (const float*)d_in[8];
  const float* res1_b  = (const float*)d_in[9];
  const float* res2_w  = (const float*)d_in[10];
  const float* res2_b  = (const float*)d_in[11];
  const float* res3_w  = (const float*)d_in[12];
  const float* res3_b  = (const float*)d_in[13];
  const float* ln1_g   = (const float*)d_in[14];
  const float* ln1_b   = (const float*)d_in[15];
  const float* ln2_g   = (const float*)d_in[16];
  const float* ln2_b   = (const float*)d_in[17];
  const float* ln3_g   = (const float*)d_in[18];
  const float* ln3_b   = (const float*)d_in[19];

  const int* row = edge;        // edge_index[0]
  const int* col = edge + EE;   // edge_index[1]

  // workspace carve-up (all 16B aligned)
  char* ws = (char*)d_ws;
  size_t off = 0;
  float* deg  = (float*)(ws + off); off += (size_t)NN * 4;
  float* dinv = (float*)(ws + off); off += (size_t)NN * 4;
  float* nrm  = (float*)(ws + off); off += (size_t)EE * 4;
  float* res  = (float*)(ws + off); off += (size_t)NN * 1024 * 4;
  float* accb = (float*)(ws + off); off += (size_t)NN * 1024 * 4;
  float* tA   = (float*)(ws + off); off += (size_t)NN * 512 * 4;
  float* tB   = (float*)(ws + off); off += (size_t)NN * 512 * 4;
  float* h1   = (float*)(ws + off); off += (size_t)NN * 256 * 4;
  float* h2   = (float*)(ws + off); off += (size_t)NN * 512 * 4;

  // graph preprocessing
  launch_zero(deg, NN, stream);
  deg_kernel<<<(EE + 255) / 256, 256, 0, stream>>>(row, deg, EE);
  dinv_kernel<<<(NN + 255) / 256, 256, 0, stream>>>(deg, dinv, NN);
  norm_kernel<<<(EE + 255) / 256, 256, 0, stream>>>(row, col, dinv, nrm, EE);

  // block 1: 128 -> 256
  run_block(x, 128, 256, cheb1_w, cheb1_b, res1_w, res1_b, ln1_g, ln1_b, h1,
            row, col, nrm, res, accb, tA, tB, stream);
  // block 2: 256 -> 512
  run_block(h1, 256, 512, cheb2_w, cheb2_b, res2_w, res2_b, ln2_g, ln2_b, h2,
            row, col, nrm, res, accb, tA, tB, stream);
  // block 3: 512 -> 1024 (writes final output)
  run_block(h2, 512, 1024, cheb3_w, cheb3_b, res3_w, res3_b, ln3_g, ln3_b,
            (float*)d_out, row, col, nrm, res, accb, tA, tB, stream);
}
